// Transformer_19713899888740
// MI455X (gfx1250) — compile-verified, matrix operand order checked by reading, not run on hardware
//
#include <hip/hip_runtime.h>

// ---------------------------------------------------------------------------
// Types for CDNA5 WMMA (wave32): v_wmma_f32_16x16x32_bf16
// ---------------------------------------------------------------------------
typedef __bf16 bf16;
typedef bf16  v16bf __attribute__((ext_vector_type(16)));
typedef bf16  v8bf  __attribute__((ext_vector_type(8)));
typedef bf16  v4bf  __attribute__((ext_vector_type(4)));
typedef float v8f   __attribute__((ext_vector_type(8)));

// model dims
#define MD_D    1024
#define MD_DFF  4096
#define MD_H    16
#define MD_L    6
#define MD_DB   768
#define MD_B    4
#define MD_S    512
#define MD_ROWS (MD_B * MD_S)   // 2048

// ---------------------------------------------------------------------------
// CDNA5 async copy global->LDS (ASYNCcnt path). LDS operand is a byte offset
// into the workgroup LDS allocation; we use a single __shared__ arena so its
// base is group-segment offset 0.
// ---------------------------------------------------------------------------
__device__ __forceinline__ void async_copy_b128(unsigned lds_byte_off, const void* gaddr) {
  asm volatile("global_load_async_to_lds_b128 %0, %1, off"
               :: "v"(lds_byte_off), "v"((unsigned long long)(uintptr_t)gaddr)
               : "memory");
}
__device__ __forceinline__ void wait_async0() {
  asm volatile("s_wait_asynccnt 0x0" ::: "memory");
}

// ---------------------------------------------------------------------------
// Operand conversion kernels (all GEMM inputs are normalized to bf16 so the
// GEMM staging is a pure byte copy via the async path):
//   cvt_flat:      f32 [..]      -> bf16 [..]        (A operands, N-major B)
//   transpose_cvt: f32 [K][N]    -> bf16 [N][K]      (K-major B: weights, V)
// ---------------------------------------------------------------------------
__global__ __launch_bounds__(256) void cvt_flat(const float* __restrict__ in,
                                                bf16* __restrict__ out, long long n4) {
  long long i = (long long)blockIdx.x * blockDim.x + threadIdx.x;
  if (i >= n4) return;
  float4 f = reinterpret_cast<const float4*>(in)[i];
  v4bf o;
  o.x = (bf16)f.x; o.y = (bf16)f.y; o.z = (bf16)f.z; o.w = (bf16)f.w;
  reinterpret_cast<v4bf*>(out)[i] = o;
}

__global__ __launch_bounds__(256) void transpose_cvt(const float* __restrict__ in,
                                                     bf16* __restrict__ out,
                                                     int K, int N) {
  __shared__ float t[32][33];
  const long long z = blockIdx.z;
  const float* ib = in  + z * (long long)K * N;
  bf16*        ob = out + z * (long long)K * N;
  const int n0 = blockIdx.x * 32, k0 = blockIdx.y * 32;
  const int tx = threadIdx.x, ty = threadIdx.y;
  #pragma unroll
  for (int i = 0; i < 32; i += 8)
    t[ty + i][tx] = ib[(long long)(k0 + ty + i) * N + n0 + tx];
  __syncthreads();
  #pragma unroll
  for (int i = 0; i < 32; i += 8)
    ob[(long long)(n0 + ty + i) * K + k0 + tx] = (bf16)t[tx][ty + i];
}

// ---------------------------------------------------------------------------
// Batched GEMM: C[z] = act( alpha * A[z] x B[z]^T + bias ),  A: [M][K] bf16,
// B: [N][K] bf16 (N-major). Block tile 256x64, 8 waves, wave tile 32x64
// -> 8x v_wmma_f32_16x16x32_bf16 per K-step. Double-buffered LDS tiles filled
// with GLOBAL_LOAD_ASYNC_TO_LDS_B128 (ASYNCcnt), one barrier per K-step.
// Fragment loads are two contiguous ds_load_b128 per lane matching the CDNA5
// 16-bit A-matrix layout (lanes 0-15: K {0..7,16..23}; 16-31: {8..15,24..31}).
// Requires M%256==0, N%64==0, K%32==0 (true for every GEMM in this model).
// ---------------------------------------------------------------------------
#define TPAD 40                     // padded row length (bf16) for 32-K tiles
#define A_TILE (256 * TPAD)         // bf16 elems per A buffer
#define B_TILE (64 * TPAD)          // bf16 elems per B buffer

struct GemmP {
  const bf16* A; const bf16* B; const float* bias; float* C;
  int K, ldc;
  long long aBatch, bBatch, cBatch, biasBatch;
  float alpha; int relu;
};

union V16u { v16bf v; v8bf h[2]; };

__global__ __launch_bounds__(256) void gemm_bf16_wmma(GemmP p) {
  __shared__ __attribute__((aligned(16))) char smem[(2 * A_TILE + 2 * B_TILE) * 2];
  bf16* const sbase = reinterpret_cast<bf16*>(smem);

  const int tid  = threadIdx.x;
  const int lane = tid & 31;
  const int wave = tid >> 5;        // 8 waves stacked along M
  const int half = lane >> 4;
  const int l16  = lane & 15;

  const long long z = blockIdx.z;
  const int bM = blockIdx.y * 256;
  const int bN = blockIdx.x * 64;
  const bf16* Ab = p.A + z * p.aBatch + (long long)bM * p.K;
  const bf16* Bb = p.B + z * p.bBatch + (long long)bN * p.K;
  float*      Cb = p.C + z * p.cBatch;

  // stage tile k0 into buffer s (async, 16B per op, EXEC all-ones)
  auto stage = [&](int s, int k0) {
    bf16* a = sbase + s * A_TILE;
    bf16* b = sbase + 2 * A_TILE + s * B_TILE;
    #pragma unroll
    for (int i = 0; i < 4; ++i) {                 // A: 256 rows x 4 octets
      int chunk = tid + (i << 8);
      int row = chunk >> 2, kq = (chunk & 3) << 3;
      async_copy_b128((unsigned)((char*)(a + row * TPAD + kq) - smem),
                      Ab + (long long)row * p.K + k0 + kq);
    }
    {                                             // B: 64 rows x 4 octets
      int row = tid >> 2, kq = (tid & 3) << 3;
      async_copy_b128((unsigned)((char*)(b + row * TPAD + kq) - smem),
                      Bb + (long long)row * p.K + k0 + kq);
    }
  };

  const v8f zero8 = {0.f,0.f,0.f,0.f,0.f,0.f,0.f,0.f};
  v8f acc[2][4] = {{zero8,zero8,zero8,zero8},{zero8,zero8,zero8,zero8}};

  const int nk = p.K >> 5;
  stage(0, 0);
  wait_async0();
  __syncthreads();

  for (int it = 0; it < nk; ++it) {
    const int cur = it & 1;
    if (it + 1 < nk) stage(cur ^ 1, (it + 1) << 5);

    const bf16* a = sbase + cur * A_TILE;
    const bf16* b = sbase + 2 * A_TILE + cur * B_TILE;

    V16u af[2], bfr[4];
    #pragma unroll
    for (int i = 0; i < 2; ++i) {
      const v8bf* pa = reinterpret_cast<const v8bf*>(a + (wave * 32 + (i << 4) + l16) * TPAD);
      af[i].h[0] = pa[half];       // K 0..7   or 8..15
      af[i].h[1] = pa[half + 2];   // K 16..23 or 24..31
    }
    #pragma unroll
    for (int j = 0; j < 4; ++j) {
      const v8bf* pb = reinterpret_cast<const v8bf*>(b + ((j << 4) + l16) * TPAD);
      bfr[j].h[0] = pb[half];
      bfr[j].h[1] = pb[half + 2];
    }
    #pragma unroll
    for (int i = 0; i < 2; ++i)
      #pragma unroll
      for (int j = 0; j < 4; ++j)
        acc[i][j] = __builtin_amdgcn_wmma_f32_16x16x32_bf16(
            false, af[i].v, false, bfr[j].v, (short)0, acc[i][j], false, false);

    wait_async0();     // next tile landed in LDS (own wave's asyncs)
    __syncthreads();   // ... and everyone else's
  }

  // epilogue: C/D layout lane L -> M = r + 8*(L>=16), N = L&15
  const float* biasB = p.bias ? (p.bias + z * p.biasBatch) : nullptr;
  #pragma unroll
  for (int i = 0; i < 2; ++i) {
    #pragma unroll
    for (int j = 0; j < 4; ++j) {
      int m0 = bM + wave * 32 + (i << 4) + (half << 3);
      int n  = bN + (j << 4) + l16;
      float bv = biasB ? biasB[n] : 0.f;
      #pragma unroll
      for (int r = 0; r < 8; ++r) {
        float v = p.alpha * acc[i][j][r] + bv;
        if (p.relu) v = fmaxf(v, 0.f);
        Cb[(long long)(m0 + r) * p.ldc + n] = v;
      }
    }
  }
}

// ---------------------------------------------------------------------------
// Row softmax over S=512 (one block per row), optional causal mask.
// ---------------------------------------------------------------------------
__global__ __launch_bounds__(256) void softmax_rows(float* sc, int S, int causal) {
  const int tid = threadIdx.x;
  const long long row = blockIdx.x;
  const int t = (int)(row % S);
  float* pr = sc + row * (long long)S;
  __shared__ float red[256];

  float v0 = pr[tid], v1 = pr[tid + 256];
  const bool m0 = causal && (tid > t);
  const bool m1 = causal && (tid + 256 > t);
  if (m0) v0 = -1e30f;
  if (m1) v1 = -1e30f;
  red[tid] = fmaxf(v0, v1);
  __syncthreads();
  for (int s = 128; s > 0; s >>= 1) {
    if (tid < s) red[tid] = fmaxf(red[tid], red[tid + s]);
    __syncthreads();
  }
  float mx = red[0];
  __syncthreads();
  float e0 = m0 ? 0.f : __expf(v0 - mx);
  float e1 = m1 ? 0.f : __expf(v1 - mx);
  red[tid] = e0 + e1;
  __syncthreads();
  for (int s = 128; s > 0; s >>= 1) {
    if (tid < s) red[tid] += red[tid + s];
    __syncthreads();
  }
  float sum = red[0];
  float inv = (sum > 0.f) ? 1.f / sum : 0.f;  // nan_to_num equivalent
  pr[tid] = e0 * inv;
  pr[tid + 256] = e1 * inv;
}

// ---------------------------------------------------------------------------
// out = LN(a + resid) * g + b ; D=1024. Values register-cached -> in-place ok.
// ---------------------------------------------------------------------------
__global__ __launch_bounds__(256) void layernorm_res(const float* a, const float* resid,
                                                     const float* g, const float* b,
                                                     float* out) {
  const int tid = threadIdx.x;
  const long long base = (long long)blockIdx.x * MD_D;
  __shared__ float red[256];
  float v[4];
  float s = 0.f;
  #pragma unroll
  for (int i = 0; i < 4; ++i) {
    int idx = tid + (i << 8);
    v[i] = a[base + idx] + (resid ? resid[base + idx] : 0.f);
    s += v[i];
  }
  red[tid] = s; __syncthreads();
  for (int k = 128; k > 0; k >>= 1) { if (tid < k) red[tid] += red[tid + k]; __syncthreads(); }
  float mean = red[0] * (1.f / MD_D);
  __syncthreads();
  float sq = 0.f;
  #pragma unroll
  for (int i = 0; i < 4; ++i) { float d = v[i] - mean; sq += d * d; }
  red[tid] = sq; __syncthreads();
  for (int k = 128; k > 0; k >>= 1) { if (tid < k) red[tid] += red[tid + k]; __syncthreads(); }
  float rstd = rsqrtf(red[0] * (1.f / MD_D) + 1e-5f);
  #pragma unroll
  for (int i = 0; i < 4; ++i) {
    int idx = tid + (i << 8);
    out[base + idx] = (v[i] - mean) * rstd * g[idx] + b[idx];
  }
}

__global__ void avg2(const float* a, const float* b, float* out, long long n) {
  long long i = (long long)blockIdx.x * blockDim.x + threadIdx.x;
  if (i < n) out[i] = 0.5f * (a[i] + b[i]);
}

// ctxh [H][ROWS][64] -> ctxc [ROWS][H*64]
__global__ void concat_heads(const float* in, float* out) {
  long long t = (long long)blockIdx.x * 256 + threadIdx.x;
  int h = (int)(t / ((long long)MD_ROWS * 64));
  long long r = t % ((long long)MD_ROWS * 64);
  int m  = (int)(r >> 6);
  int kk = (int)(r & 63);
  out[(long long)m * MD_D + h * 64 + kk] = in[t];
}

// ---------------------------------------------------------------------------
// Host-side orchestration
// ---------------------------------------------------------------------------
enum BLayout { B_KN, B_NK };   // source layout of B in f32 memory

static void gemm_full(hipStream_t st,
                      const float* A, int batchedA,
                      const float* B, BLayout blay, int batchedB,
                      const float* bias, long long biasB,
                      float* C, int ldc, long long cB,
                      int M, int N, int K, int batches, float alpha, int relu,
                      bf16* Abf, bf16* Bbf) {
  const long long aspan = (long long)M * K * (batchedA ? batches : 1);
  cvt_flat<<<(int)((aspan / 4 + 255) / 256), 256, 0, st>>>(A, Abf, aspan / 4);
  const long long bper = (long long)N * K;
  if (blay == B_NK) {
    const long long bspan = bper * (batchedB ? batches : 1);
    cvt_flat<<<(int)((bspan / 4 + 255) / 256), 256, 0, st>>>(B, Bbf, bspan / 4);
  } else {
    dim3 g(N / 32, K / 32, batchedB ? batches : 1);
    transpose_cvt<<<g, dim3(32, 8), 0, st>>>(B, Bbf, K, N);
  }
  GemmP p;
  p.A = Abf; p.B = Bbf; p.bias = bias; p.C = C;
  p.K = K; p.ldc = ldc;
  p.aBatch = batchedA ? (long long)M * K : 0;
  p.bBatch = batchedB ? bper : 0;
  p.cBatch = cB; p.biasBatch = biasB;
  p.alpha = alpha; p.relu = relu;
  dim3 grid(N / 64, M / 256, batches);
  gemm_bf16_wmma<<<grid, 256, 0, st>>>(p);
}

struct AttnW { const float *wq,*bq,*wk,*bk,*wv,*bv,*woh,*boh,*wo,*bo; int dq, dkv; };

// q/k/v/ctx layout: [H][B*S][64]
static void run_mha(hipStream_t st, const AttnW& a, int l,
                    const float* qx, const float* kvx, int causal, float* outp,
                    float* QB, float* KB, float* VB, float* SC,
                    float* CT, float* CH, float* CC, bf16* Abf, bf16* Bbf) {
  const float* wq  = a.wq  + (long long)l * MD_H * a.dq  * 64;
  const float* bq  = a.bq  + (long long)l * MD_H * 64;
  const float* wk  = a.wk  + (long long)l * MD_H * a.dkv * 64;
  const float* bk  = a.bk  + (long long)l * MD_H * 64;
  const float* wv  = a.wv  + (long long)l * MD_H * a.dkv * 64;
  const float* bv  = a.bv  + (long long)l * MD_H * 64;
  const float* woh = a.woh + (long long)l * MD_H * 64 * 64;
  const float* boh = a.boh + (long long)l * MD_H * 64;
  const float* wo  = a.wo  + (long long)l * MD_D * MD_D;
  const float* bo  = a.bo  + (long long)l * MD_D;

  const long long HB = (long long)MD_ROWS * 64;  // per-head block

  // Q/K/V projections (batched over heads): A shared, B = weights [K][64]
  gemm_full(st, qx,  0, wq, B_KN, 1, bq, 64, QB, 64, HB,
            MD_ROWS, 64, a.dq,  MD_H, 1.f, 0, Abf, Bbf);
  gemm_full(st, kvx, 0, wk, B_KN, 1, bk, 64, KB, 64, HB,
            MD_ROWS, 64, a.dkv, MD_H, 1.f, 0, Abf, Bbf);
  gemm_full(st, kvx, 0, wv, B_KN, 1, bv, 64, VB, 64, HB,
            MD_ROWS, 64, a.dkv, MD_H, 1.f, 0, Abf, Bbf);

  // scores = (1/8) Q K^T : B = K is already [s][64] = [N][K]
  gemm_full(st, QB, 1, KB, B_NK, 1, nullptr, 0, SC, MD_S, (long long)MD_S * MD_S,
            MD_S, MD_S, 64, MD_H * MD_B, 0.125f, 0, Abf, Bbf);
  softmax_rows<<<MD_H * MD_B * MD_S, 256, 0, st>>>(SC, MD_S, causal);

  // ctx = P V : B = V is [s][64] = [K][N] -> transpose
  gemm_full(st, SC, 1, VB, B_KN, 1, nullptr, 0, CT, 64, (long long)MD_S * 64,
            MD_S, 64, MD_S, MD_H * MD_B, 1.f, 0, Abf, Bbf);

  // per-head output projection (64x64)
  gemm_full(st, CT, 1, woh, B_KN, 1, boh, 64, CH, 64, HB,
            MD_ROWS, 64, 64, MD_H, 1.f, 0, Abf, Bbf);

  // concat heads, then final D x D projection
  concat_heads<<<(MD_H * MD_ROWS * 64) / 256, 256, 0, st>>>(CH, CC);
  gemm_full(st, CC, 0, wo, B_KN, 0, bo, 0, outp, MD_D, 0,
            MD_ROWS, MD_D, MD_D, 1, 1.f, 0, Abf, Bbf);
}

static void run_ffn(hipStream_t st, const float* x,
                    const float* w1, const float* b1,
                    const float* w2, const float* b2,
                    float* FH, float* outp, bf16* Abf, bf16* Bbf) {
  gemm_full(st, x,  0, w1, B_KN, 0, b1, 0, FH,   MD_DFF, 0,
            MD_ROWS, MD_DFF, MD_D,   1, 1.f, 1, Abf, Bbf);   // relu
  gemm_full(st, FH, 0, w2, B_KN, 0, b2, 0, outp, MD_D,   0,
            MD_ROWS, MD_D,   MD_DFF, 1, 1.f, 0, Abf, Bbf);
}

extern "C" void kernel_launch(void* const* d_in, const int* in_sizes, int n_in,
                              void* d_out, int out_size, void* d_ws, size_t ws_size,
                              hipStream_t stream) {
  (void)in_sizes; (void)n_in; (void)out_size; (void)ws_size;

  const float* src  = (const float*)d_in[0];
  const float* tgt  = (const float*)d_in[1];
  const float* bert = (const float*)d_in[2];
  // d_in[3..5]: padding masks (all false in the reference) -> ignored.

  const int P = 6;   // params leaves: jax pytree (sorted-key) flatten order
  auto F = [&](int i) { return (const float*)d_in[P + i]; };
  auto mkAttn = [&](int base, int dq, int dkv) {
    AttnW a;
    a.bk = F(base+0); a.bo = F(base+1); a.boh = F(base+2); a.bq = F(base+3);
    a.bv = F(base+4); a.wk = F(base+5); a.wo = F(base+6); a.woh = F(base+7);
    a.wq = F(base+8); a.wv = F(base+9); a.dq = dq; a.dkv = dkv;
    return a;
  };

  AttnW dec_bda = mkAttn(0,  MD_D, MD_DB);
  AttnW dec_eda = mkAttn(10, MD_D, MD_D);
  const float* dec_ffb1 = F(20); const float* dec_ffb2 = F(21);
  const float* dec_ffw1 = F(22); const float* dec_ffw2 = F(23);
  const float* dec_ln1b = F(24); const float* dec_ln1g = F(25);
  const float* dec_ln2b = F(26); const float* dec_ln2g = F(27);
  const float* dec_ln3b = F(28); const float* dec_ln3g = F(29);
  AttnW dec_sa  = mkAttn(30, MD_D, MD_D);
  AttnW enc_ba  = mkAttn(40, MD_D, MD_DB);
  const float* enc_ffb1 = F(50); const float* enc_ffb2 = F(51);
  const float* enc_ffw1 = F(52); const float* enc_ffw2 = F(53);
  const float* enc_ln1b = F(54); const float* enc_ln1g = F(55);
  const float* enc_ln2b = F(56); const float* enc_ln2g = F(57);
  AttnW enc_sa  = mkAttn(58, MD_D, MD_D);

  // workspace arena (float units)
  const long long M2 = (long long)MD_ROWS * MD_D;  // 2M floats
  float* W   = (float*)d_ws;
  float* X   = W;             // encoder activations / enc_out
  float* Y   = X   + M2;      // decoder activations
  float* QB  = Y   + M2;
  float* KB  = QB  + M2;
  float* VB  = KB  + M2;
  float* CT  = VB  + M2;
  float* CH  = CT  + M2;
  float* CC  = CH  + M2;
  float* A1  = CC  + M2;
  float* A2  = A1  + M2;
  float* SAb = A2  + M2;
  float* FH  = SAb + M2;                               // 2048 x 4096
  float* SC  = FH  + (long long)MD_ROWS * MD_DFF;      // 64 x 512 x 512
  float* END = SC  + (long long)MD_H * MD_B * MD_S * MD_S;
  bf16*  Abf = (bf16*)END;                             // 16.7M bf16
  bf16*  Bbf = Abf + (long long)MD_H * MD_B * MD_S * MD_S;  // 4M bf16
  // Bbf end: + MD_D*MD_DFF

  hipMemcpyAsync(X, src, M2 * sizeof(float), hipMemcpyDeviceToDevice, stream);
  hipMemcpyAsync(Y, tgt, M2 * sizeof(float), hipMemcpyDeviceToDevice, stream);

  // ---------------- encoder ----------------
  for (int l = 0; l < MD_L; ++l) {
    run_mha(stream, enc_sa, l, X, X,    0, A1, QB, KB, VB, SC, CT, CH, CC, Abf, Bbf);
    run_mha(stream, enc_ba, l, X, bert, 0, A2, QB, KB, VB, SC, CT, CH, CC, Abf, Bbf);
    avg2<<<(int)(M2 / 256), 256, 0, stream>>>(A1, A2, A1, M2);
    layernorm_res<<<MD_ROWS, 256, 0, stream>>>(A1, X,
        enc_ln1g + (long long)l * MD_D, enc_ln1b + (long long)l * MD_D, X);
    run_ffn(stream, X,
            enc_ffw1 + (long long)l * MD_D * MD_DFF, enc_ffb1 + (long long)l * MD_DFF,
            enc_ffw2 + (long long)l * MD_DFF * MD_D, enc_ffb2 + (long long)l * MD_D,
            FH, A1, Abf, Bbf);
    layernorm_res<<<MD_ROWS, 256, 0, stream>>>(A1, X,
        enc_ln2g + (long long)l * MD_D, enc_ln2b + (long long)l * MD_D, X);
  }

  // ---------------- decoder ----------------
  for (int l = 0; l < MD_L; ++l) {
    run_mha(stream, dec_sa, l, Y, Y, 1, A1, QB, KB, VB, SC, CT, CH, CC, Abf, Bbf);
    layernorm_res<<<MD_ROWS, 256, 0, stream>>>(A1, Y,
        dec_ln1g + (long long)l * MD_D, dec_ln1b + (long long)l * MD_D, SAb);
    run_mha(stream, dec_eda, l, SAb, X,    0, A1, QB, KB, VB, SC, CT, CH, CC, Abf, Bbf);
    run_mha(stream, dec_bda, l, SAb, bert, 0, A2, QB, KB, VB, SC, CT, CH, CC, Abf, Bbf);
    avg2<<<(int)(M2 / 256), 256, 0, stream>>>(A1, A2, A1, M2);
    layernorm_res<<<MD_ROWS, 256, 0, stream>>>(A1, SAb,
        dec_ln2g + (long long)l * MD_D, dec_ln2b + (long long)l * MD_D, Y);
    run_ffn(stream, Y,
            dec_ffw1 + (long long)l * MD_D * MD_DFF, dec_ffb1 + (long long)l * MD_DFF,
            dec_ffw2 + (long long)l * MD_DFF * MD_D, dec_ffb2 + (long long)l * MD_D,
            FH, A1, Abf, Bbf);
    layernorm_res<<<MD_ROWS, 256, 0, stream>>>(A1, Y,
        dec_ln3g + (long long)l * MD_D, dec_ln3b + (long long)l * MD_D, Y);
  }

  hipMemcpyAsync(d_out, Y, M2 * sizeof(float), hipMemcpyDeviceToDevice, stream);
}
